// GNNModel_76690936037952
// MI455X (gfx1250) — matrix operand order repaired and checked by reading, not compiled, not run on hardware
//
#include <hip/hip_runtime.h>
#include <hip/hip_bf16.h>
#include <cstdint>

// ---------------------------------------------------------------------------
// Problem constants (from the reference)
// ---------------------------------------------------------------------------
#define NB      1024          // boards
#define NODES   64
#define INF_    15            // input embedding
#define HH      4             // heads
#define CC      64            // channels per head
#define HC      256           // H*C
#define NTOT    (NB*NODES)    // 65536 nodes
#define EPER    512           // edges per board (before self loops)
#define ETOT    (NB*EPER)     // 524288
#define EBOARD  (EPER+NODES)  // 576 edges incl. self loops
#define K1      16384         // NODES*H*C
#define N1      4096          // NODES*C
#define OUTC    10

typedef __attribute__((ext_vector_type(16))) _Float16     v16h;
typedef __attribute__((ext_vector_type(8)))  float        v8f;
typedef __attribute__((ext_vector_type(4)))  unsigned int u32x4;
typedef __attribute__((ext_vector_type(8)))  int          i32x8;
typedef __attribute__((ext_vector_type(4)))  int          i32x4;

#if defined(__has_builtin)
#  if __has_builtin(__builtin_amdgcn_tensor_load_to_lds)
#    define HAVE_TDM 1
#  endif
#endif
#ifndef HAVE_TDM
#  define HAVE_TDM 0
#endif

// ---------------------------------------------------------------------------
// WMMA fragment helpers (wave32; A/B use this code's contiguous K order)
//   A 16x32 f16 : lane L holds row M=L&15; elem j -> K=j+(L>=16?16:0)
//   B 32x16 f16 : lane L holds col N=L&15; elem j -> K=j+(L>=16?16:0)
//   C/D 16x16 f32: lane L col N=L&15; vgpr r -> M=r+(L>=16?8:0)
// ---------------------------------------------------------------------------
__device__ __forceinline__ v16h load_a_frag_g(const float* __restrict__ A,
                                              int row0, int lda, int k0, int lane) {
  const int m  = row0 + (lane & 15);
  const int lo = (lane & 16) ? 16 : 0;
  const float* base = A + (size_t)m * lda + k0;
  v16h a;
#pragma unroll
  for (int p = 0; p < 8; ++p) {
    const int kk = 2 * p + lo;   // pairs are K-contiguous
    const float2 v = *(const float2*)(base + kk);
    a[2 * p]     = (_Float16)v.x;
    a[2 * p + 1] = (_Float16)v.y;
  }
  return a;
}

__device__ __forceinline__ v16h load_b_frag_g(const float* __restrict__ Bm,
                                              int col0, int ldb, int k0, int lane) {
  const int n  = col0 + (lane & 15);
  const int lo = (lane & 16) ? 16 : 0;
  const float* base = Bm + (size_t)(k0 + lo) * ldb + n;
  v16h b;
#pragma unroll
  for (int j = 0; j < 16; ++j) b[j] = (_Float16)base[(size_t)j * ldb];
  return b;
}

__device__ __forceinline__ void store_tile(float* __restrict__ D, int row0, int col0,
                                           int ldd, int lane, v8f c,
                                           const float* __restrict__ bias, bool relu) {
  const int n  = col0 + (lane & 15);
  const int ro = (lane & 16) ? 8 : 0;
#pragma unroll
  for (int r = 0; r < 8; ++r) {
    float v = c[r];
    if (bias) v += bias[n];
    if (relu) v = fmaxf(v, 0.f);
    D[(size_t)(row0 + r + ro) * ldd + n] = v;
  }
}

// ---------------------------------------------------------------------------
// TDM: 2D tile (f32) of a 2D row-major tensor -> LDS, per CDNA5 ISA D# layout
//   group0: [1:0]=count=1 | lds_addr[63:32] | global_addr[120:64] | type=2
//   group1: data_size=4B | tensor_dim0/1 | tile_dim0/1 | tensor_dim0_stride
// ---------------------------------------------------------------------------
#if HAVE_TDM
__device__ __forceinline__ void tdm_load_2d(const float* gsrc, const float* ldst,
                                            unsigned tensor_w, unsigned tensor_h,
                                            unsigned tile_w, unsigned tile_h) {
  const unsigned long long ga = (unsigned long long)(uintptr_t)gsrc;
  const unsigned lds = (unsigned)(uintptr_t)ldst;   // low 32 bits = LDS offset
  u32x4 g0;
  g0[0] = 1u;                                                   // count=1
  g0[1] = lds;                                                  // lds_addr
  g0[2] = (unsigned)ga;                                         // global[31:0]
  g0[3] = ((unsigned)(ga >> 32) & 0x01FFFFFFu) | (2u << 30);    // [56:32]|type=2
  i32x8 g1;
  g1[0] = (int)(2u << 16);                                      // data_size=4B
  g1[1] = (int)((tensor_w & 0xFFFFu) << 16);                    // dim0 lo16
  g1[2] = (int)(((tensor_w >> 16) & 0xFFFFu) |
                ((tensor_h & 0xFFFFu) << 16));                  // dim0 hi|dim1 lo
  g1[3] = (int)(((tensor_h >> 16) & 0xFFFFu) | (tile_w << 16)); // dim1 hi|tile0
  g1[4] = (int)(tile_h & 0xFFFFu);                              // tile_dim1
  g1[5] = (int)tensor_w;                                        // dim0_stride lo32
  g1[6] = 0;
  g1[7] = 0;
  const i32x4 z4 = {0, 0, 0, 0};
#if __clang_major__ >= 23
  const i32x8 z8 = {0, 0, 0, 0, 0, 0, 0, 0};
  __builtin_amdgcn_tensor_load_to_lds(g0, g1, z4, z4, z8, 0);
#else
  __builtin_amdgcn_tensor_load_to_lds(g0, g1, z4, z4, 0);
#endif
}
#endif

// ---------------------------------------------------------------------------
// Kernel 1: y0 = relu(x @ w0 + b0)     [N,15]x[15,64] -> [N,64]
// ---------------------------------------------------------------------------
__global__ void k_lin0(const float* __restrict__ x, const float* __restrict__ w0,
                       const float* __restrict__ b0, float* __restrict__ y0) {
  const int t = blockIdx.x * blockDim.x + threadIdx.x;   // N*64 threads
  const int n = t >> 6, c = t & 63;
  const float* xr = x + (size_t)n * INF_;
  float acc = b0[c];
#pragma unroll
  for (int k = 0; k < INF_; ++k) acc = fmaf(xr[k], w0[k * CC + c], acc);
  y0[(size_t)n * CC + c] = fmaxf(acc, 0.f);
}

// ---------------------------------------------------------------------------
// Kernel 2: xl = y0 @ wl ; xr = y0 @ wr    [N,64]x[64,256]  (blockIdx.y picks)
// One wave per 16x64 strip: A fragment reused across 4 WMMAs per K chunk.
// ---------------------------------------------------------------------------
__global__ void k_gemm_lr(const float* __restrict__ Y, const float* __restrict__ Wl,
                          const float* __restrict__ Wr, float* __restrict__ XL,
                          float* __restrict__ XR) {
  const float* W = blockIdx.y ? Wr : Wl;
  float*       O = blockIdx.y ? XR : XL;
  const int wave = (blockIdx.x * blockDim.x + threadIdx.x) >> 5;  // 16384 waves
  const int lane = threadIdx.x & 31;
  const int row0 = (wave >> 2) * 16;            // 4096 row tiles
  const int col0 = (wave & 3) * 64;             // 4 col groups of 64
  v8f acc[4] = {};
#pragma unroll
  for (int k0 = 0; k0 < CC; k0 += 32) {
    const v16h a = load_a_frag_g(Y, row0, CC, k0, lane);
#pragma unroll
    for (int t = 0; t < 4; ++t) {
      const v16h b = load_b_frag_g(W, col0 + 16 * t, HC, k0, lane);
      acc[t] = __builtin_amdgcn_wmma_f32_16x16x32_f16(false, a, false, b,
                                                      (short)0, acc[t], false, false);
    }
  }
#pragma unroll
  for (int t = 0; t < 4; ++t)
    store_tile(O, row0, col0 + 16 * t, HC, lane, acc[t], nullptr, false);
}

// ---------------------------------------------------------------------------
// Kernel 3: GATv2 edge stage, one block per board, all in LDS (no atomics).
// Thread t owns (dst d = t>>2, head h = t&3): segment max / sum / accumulate.
// ---------------------------------------------------------------------------
__global__ void k_gat(const float* __restrict__ XL, const float* __restrict__ XR,
                      const int* __restrict__ esrc, const int* __restrict__ edst,
                      const float* __restrict__ att, const float* __restrict__ gbias,
                      float* __restrict__ Y1) {
  extern __shared__ float smem[];
  float* s_xl    = smem;                    // 64*256
  float* s_xr    = s_xl + NODES * HC;       // 64*256
  float* s_logit = s_xr + NODES * HC;       // 576*4
  float* s_att   = s_logit + EBOARD * HH;   // 256
  int*   s_src   = (int*)(s_att + HC);      // 576
  int*   s_dst   = s_src + EBOARD;          // 576

  const int tid = threadIdx.x;
  const int board = blockIdx.x;
  const int nodebase = board * NODES;
  const float* xlb = XL + (size_t)nodebase * HC;
  const float* xrb = XR + (size_t)nodebase * HC;

  for (int i = tid; i < NODES * HC / 4; i += blockDim.x) {
    ((float4*)s_xl)[i] = ((const float4*)xlb)[i];
    ((float4*)s_xr)[i] = ((const float4*)xrb)[i];
  }
  s_att[tid] = att[tid];                                   // 256 == blockDim
  for (int e = tid; e < EPER; e += blockDim.x) {
    s_src[e] = esrc[board * EPER + e] & (NODES - 1);       // local ids
    s_dst[e] = edst[board * EPER + e] & (NODES - 1);
  }
  if (tid < NODES) { s_src[EPER + tid] = tid; s_dst[EPER + tid] = tid; }
  __syncthreads();

  // logits[e][h] = sum_c leaky(xl[s]+xr[d]) * att[h][c]
  for (int p = tid; p < EBOARD * HH; p += blockDim.x) {
    const int e = p >> 2, h = p & 3;
    const float* pl = s_xl + s_src[e] * HC + h * CC;
    const float* pr = s_xr + s_dst[e] * HC + h * CC;
    const float* pa = s_att + h * CC;
    float acc = 0.f;
#pragma unroll 8
    for (int c = 0; c < CC; ++c) {
      float v = pl[c] + pr[c];
      v = v > 0.f ? v : 0.2f * v;                          // LeakyReLU
      acc = fmaf(v, pa[c], acc);
    }
    s_logit[e * HH + h] = acc;
  }
  __syncthreads();

  // segment softmax + weighted aggregation, one thread per (dst, head)
  const int d = tid >> 2, h = tid & 3;
  float m = -3.4e38f;
  for (int e = 0; e < EBOARD; ++e)
    if (s_dst[e] == d) m = fmaxf(m, s_logit[e * HH + h]);
  float den = 0.f;
  for (int e = 0; e < EBOARD; ++e)
    if (s_dst[e] == d) den += __expf(s_logit[e * HH + h] - m);
  const float inv = 1.f / den;

  float acc[CC];
#pragma unroll
  for (int c = 0; c < CC; ++c) acc[c] = 0.f;
  for (int e = 0; e < EBOARD; ++e) {
    if (s_dst[e] != d) continue;
    const float alpha = __expf(s_logit[e * HH + h] - m) * inv;
    const float* pl = s_xl + s_src[e] * HC + h * CC;
#pragma unroll
    for (int c = 0; c < CC; ++c) acc[c] = fmaf(alpha, pl[c], acc[c]);
  }
  float* o = Y1 + (size_t)(nodebase + d) * HC + h * CC;
  const float* gb = gbias + h * CC;
#pragma unroll
  for (int c = 0; c < CC; ++c) o[c] = acc[c] + gb[c];
}

// ---------------------------------------------------------------------------
// Kernel 4: o1 = relu(y1 @ w1 + b1)   [1024,16384]x[16384,4096]  (137 GFLOP)
// Block = 8 waves (2x4) -> 64x128 tile. K chunks (32) double-buffered in LDS,
// filled by the Tensor Data Mover (wave 0 issues descriptors, s_wait_tensorcnt)
// or cooperative float4 staging when TDM builtin is unavailable.
// Each wave: 1 A frag + 4 B frags -> 4 WMMAs per chunk.
// ---------------------------------------------------------------------------
__global__ void k_lin1(const float* __restrict__ Y1, const float* __restrict__ W1,
                       const float* __restrict__ B1, float* __restrict__ O) {
  __shared__ float s_a[2][64 * 32];    // [buf][row][k]   16 KB
  __shared__ float s_b[2][32 * 128];   // [buf][k][col]   32 KB
  const int tid  = threadIdx.x;
  const int lane = tid & 31, warp = tid >> 5;
  const int bm = blockIdx.x >> 5;                 // 16 M-blocks (1024/64)
  const int bn = blockIdx.x & 31;                 // 32 N-blocks (4096/128)
  const int row0 = bm * 64, col0 = bn * 128;
  const int wm = (warp >> 1) * 16;                // 0,16,32,48
  const int wn = (warp & 1) * 64;                 // 0,64
  v8f acc[4] = {};

  auto load_panels = [&](int k0, int buf) {
#if HAVE_TDM
    if (warp == 0) {   // one wave drives the DMA engine
      tdm_load_2d(Y1 + (size_t)row0 * K1 + k0, &s_a[buf][0], K1, NB, 32, 64);
      tdm_load_2d(W1 + (size_t)k0 * N1 + col0, &s_b[buf][0], N1, K1, 128, 32);
    }
#else
    {  // A: 64x32 f32, each thread two float4
      const int i = tid * 8;
      const int r = i >> 5, kk = i & 31;
      const float* gp = Y1 + (size_t)(row0 + r) * K1 + k0 + kk;
      *(float4*)(&s_a[buf][r * 32 + kk])     = *(const float4*)gp;
      *(float4*)(&s_a[buf][r * 32 + kk + 4]) = *(const float4*)(gp + 4);
    }
    {  // B: 32x128 f32, each thread four float4
      const int i  = tid * 16;
      const int kr = i >> 7, cp = i & 127;
      const float* gp = W1 + (size_t)(k0 + kr) * N1 + col0 + cp;
      float* lp = &s_b[buf][kr * 128 + cp];
#pragma unroll
      for (int q = 0; q < 4; ++q)
        *(float4*)(lp + 4 * q) = *(const float4*)(gp + 4 * q);
    }
#endif
  };

  load_panels(0, 0);
#if HAVE_TDM
  if (warp == 0) __builtin_amdgcn_s_wait_tensorcnt(0);
#endif
  __syncthreads();

  for (int k0 = 0; k0 < K1; k0 += 32) {
    const int bufc = (k0 >> 5) & 1;
    if (k0 + 32 < K1) load_panels(k0 + 32, bufc ^ 1);   // overlap next DMA

    // compute current chunk
    const float* pa = &s_a[bufc][0];
    const float* pb = &s_b[bufc][0];
    v16h a;
    {
      const int m  = wm + (lane & 15);
      const int lo = (lane & 16) ? 16 : 0;
#pragma unroll
      for (int j = 0; j < 16; ++j) {
        const int kk = j + lo;
        a[j] = (_Float16)pa[m * 32 + kk];
      }
    }
    const int lo2 = (lane & 16) ? 16 : 0;
#pragma unroll
    for (int t = 0; t < 4; ++t) {
      const int n = wn + t * 16 + (lane & 15);
      v16h b;
#pragma unroll
      for (int j = 0; j < 16; ++j) b[j] = (_Float16)pb[(j + lo2) * 128 + n];
      acc[t] = __builtin_amdgcn_wmma_f32_16x16x32_f16(false, a, false, b,
                                                      (short)0, acc[t], false, false);
    }

#if HAVE_TDM
    if (warp == 0) __builtin_amdgcn_s_wait_tensorcnt(0);  // next buffers landed
#endif
    __syncthreads();
  }

#pragma unroll
  for (int t = 0; t < 4; ++t)
    store_tile(O, row0 + wm, col0 + wn + 16 * t, N1, lane, acc[t], B1, true);
}

// ---------------------------------------------------------------------------
// Kernel 5: o2 = relu(o1 @ w2 + b2)   [1024,4096]x[4096,64]
// One wave per 16x64 strip (A fragment reused across 4 WMMAs), K loop 128.
// ---------------------------------------------------------------------------
__global__ void k_lin2(const float* __restrict__ O1, const float* __restrict__ W2,
                       const float* __restrict__ B2, float* __restrict__ O2) {
  const int wave = (blockIdx.x * blockDim.x + threadIdx.x) >> 5;   // 64 waves
  const int lane = threadIdx.x & 31;
  const int row0 = wave * 16;
  v8f acc[4] = {};
  for (int k0 = 0; k0 < N1; k0 += 32) {
    if (k0 + 32 < N1)
      __builtin_prefetch(W2 + (size_t)(k0 + 32 + (lane >> 1)) * CC + (lane & 1) * 32, 0, 1);
    const v16h a = load_a_frag_g(O1, row0, N1, k0, lane);
#pragma unroll
    for (int t = 0; t < 4; ++t) {
      const v16h b = load_b_frag_g(W2, 16 * t, CC, k0, lane);
      acc[t] = __builtin_amdgcn_wmma_f32_16x16x32_f16(false, a, false, b,
                                                      (short)0, acc[t], false, false);
    }
  }
#pragma unroll
  for (int t = 0; t < 4; ++t)
    store_tile(O2, row0, 16 * t, CC, lane, acc[t], B2, true);
}

// ---------------------------------------------------------------------------
// Kernel 6: out = o2 @ w3 + b3   [1024,64]x[64,10]  (tiny)
// ---------------------------------------------------------------------------
__global__ void k_lin3(const float* __restrict__ O2, const float* __restrict__ W3,
                       const float* __restrict__ B3, float* __restrict__ out) {
  const int t = blockIdx.x * blockDim.x + threadIdx.x;
  if (t >= NB * OUTC) return;
  const int n = t / OUTC, o = t % OUTC;
  float acc = B3[o];
#pragma unroll
  for (int c = 0; c < CC; ++c) acc = fmaf(O2[(size_t)n * CC + c], W3[c * OUTC + o], acc);
  out[t] = acc;
}

// ---------------------------------------------------------------------------
// Host-side launcher
// ---------------------------------------------------------------------------
extern "C" void kernel_launch(void* const* d_in, const int* in_sizes, int n_in,
                              void* d_out, int out_size, void* d_ws, size_t ws_size,
                              hipStream_t stream) {
  const float* x     = (const float*)d_in[0];
  const float* w0    = (const float*)d_in[1];
  const float* b0    = (const float*)d_in[2];
  const float* wl    = (const float*)d_in[3];
  const float* wr    = (const float*)d_in[4];
  const float* att   = (const float*)d_in[5];
  const float* gbias = (const float*)d_in[6];
  const float* w1    = (const float*)d_in[7];
  const float* b1    = (const float*)d_in[8];
  const float* w2    = (const float*)d_in[9];
  const float* b2    = (const float*)d_in[10];
  const float* w3    = (const float*)d_in[11];
  const float* b3    = (const float*)d_in[12];
  const int*   eidx  = (const int*)d_in[13];
  const int* esrc = eidx;
  const int* edst = eidx + ETOT;

  // workspace layout (bytes), with reuse of dead buffers
  char* ws = (char*)d_ws;
  float* y0 = (float*)(ws);                           //  16 MiB [N,64]
  float* xl = (float*)(ws + (size_t)16777216);        //  64 MiB [N,256]
  float* xr = (float*)(ws + (size_t)83886080);        //  64 MiB [N,256]
  float* y1 = (float*)(ws + (size_t)150994944);       //  64 MiB [N,256]
  float* o1 = (float*)(ws);                           //  reuse y0: [1024,4096]
  float* o2 = (float*)(ws + (size_t)16777216);        //  reuse xl: [1024,64]

  k_lin0<<<(NTOT * CC) / 256, 256, 0, stream>>>(x, w0, b0, y0);
  k_gemm_lr<<<dim3(2048, 2), 256, 0, stream>>>(y0, wl, wr, xl, xr);

  const size_t gat_smem =
      (size_t)(NODES * HC * 2 + EBOARD * HH + HC) * sizeof(float) +
      (size_t)(EBOARD * 2) * sizeof(int);             // ~146 KB of 320 KB LDS
  k_gat<<<NB, 256, gat_smem, stream>>>(xl, xr, esrc, edst, att, gbias, y1);

  k_lin1<<<512, 256, 0, stream>>>(y1, w1, b1, o1);    // 16 x 32 block tiles
  k_lin2<<<8, 256, 0, stream>>>(o1, w2, b2, o2);
  k_lin3<<<(NB * OUTC + 255) / 256, 256, 0, stream>>>(o2, w3, b3, (float*)d_out);
}
